// SoftmaxSelfAttention_3040836845898
// MI455X (gfx1250) — matrix operand order repaired and checked by reading, not compiled, not run on hardware
//
#include <hip/hip_runtime.h>

// -------- CDNA5 WMMA types (gfx1250, wave32) --------
typedef __attribute__((ext_vector_type(16))) __bf16 v16bf;
typedef __attribute__((ext_vector_type(8)))  float  v8f;
typedef unsigned short ushort_t;

#define SEQ     2048
#define HD      64
#define BR      128     // Q rows per workgroup
#define BC      64      // K/V rows per tile iteration
#define WAVES   8
#define THREADS (WAVES * 32)
#define KSTR    72      // padded LDS row stride (bf16 elems): 144B = 9*16B
#define PSTR    72      // padded sP row stride (f32 elems)

union Frag16 {              // 16x32 bf16 A-frag or 32x16 bf16 B-frag (8 VGPRs)
    v16bf          v;
    unsigned short u[16];
    unsigned int   w[8];
    uint4          q[2];
};
union Frag8f {              // 16x16 f32 C/D frag (8 VGPRs)
    v8f   v;
    float f[8];
};

__device__ __forceinline__ unsigned short f2bf(float f) {
    unsigned int u = __builtin_bit_cast(unsigned int, f);
    u += 0x7FFFu + ((u >> 16) & 1u);          // round-to-nearest-even
    return (unsigned short)(u >> 16);
}

#if __has_builtin(__builtin_amdgcn_cvt_pk_bf16_f32)
__device__ __forceinline__ unsigned int f2bf2(float a, float b) {
    return __builtin_bit_cast(unsigned int, __builtin_amdgcn_cvt_pk_bf16_f32(a, b));
}
#else
__device__ __forceinline__ unsigned int f2bf2(float a, float b) {
    return (unsigned int)f2bf(a) | ((unsigned int)f2bf(b) << 16);
}
#endif

// ---- CDNA5 async global->LDS DMA (ASYNCcnt-tracked, no VGPR round trip) ----
__device__ __forceinline__ void async_cp_b128(const void* g, const void* lds) {
    const unsigned           l  = (unsigned)(uintptr_t)lds;   // LDS addr = low 32 bits
    const unsigned long long ga = (unsigned long long)(uintptr_t)g;
    asm volatile("global_load_async_to_lds_b128 %0, %1, off"
                 :: "v"(l), "v"(ga) : "memory");
}
__device__ __forceinline__ void wait_async0() {
    asm volatile("s_wait_asynccnt 0x0" ::: "memory");
}

// 16-lane max reduction in pure DPP VALU (no ds_bpermute)
template <int CTRL>
__device__ __forceinline__ float dpp_max_step(float x) {
    int y = __builtin_amdgcn_mov_dpp(__builtin_bit_cast(int, x), CTRL, 0xf, 0xf, true);
    return fmaxf(x, __builtin_bit_cast(float, y));
}
__device__ __forceinline__ float row16_max(float x) {
    x = dpp_max_step<0xB1>(x);    // quad_perm [1,0,3,2]  (xor 1)
    x = dpp_max_step<0x4E>(x);    // quad_perm [2,3,0,1]  (xor 2)
    x = dpp_max_step<0x124>(x);   // row_ror:4
    x = dpp_max_step<0x128>(x);   // row_ror:8
    return x;
}

// ================= pre-pass: fp32 -> bf16 (Q scaled, V transposed) =============
__global__ __launch_bounds__(THREADS)
void convert_qkv(const float* __restrict__ Q, const float* __restrict__ K,
                 const float* __restrict__ V,
                 ushort_t* __restrict__ wQ,   // [bh][seq][hd] bf16, pre-scaled
                 ushort_t* __restrict__ wK,   // [bh][seq][hd] bf16
                 ushort_t* __restrict__ wVt)  // [bh][hd][seq] bf16 (transposed)
{
    __shared__ __align__(16) ushort_t sT[HD * KSTR];   // transpose staging

    const int tid = threadIdx.x;
    const int bh  = blockIdx.x / (SEQ / BC);
    const int kv0 = (blockIdx.x % (SEQ / BC)) * BC;

    const size_t base = (size_t)bh * SEQ * HD;
    const float4* Q4 = (const float4*)(Q + base + (size_t)kv0 * HD);
    const float4* K4 = (const float4*)(K + base + (size_t)kv0 * HD);
    const float4* V4 = (const float4*)(V + base + (size_t)kv0 * HD);
    ushort_t* wQs = wQ + base + (size_t)kv0 * HD;
    ushort_t* wKs = wK + base + (size_t)kv0 * HD;

    const float qscale = 0.125f * 1.44269504088896340736f; // 1/sqrt(64)*log2(e)

    #pragma unroll
    for (int j = 0; j < (BC * HD) / (4 * THREADS); ++j) {   // 4 iters
        const int idx = tid + j * THREADS;
        const int n   = idx >> 4;
        const int d4  = (idx & 15) * 4;
        const float4 qq = Q4[idx];
        const float4 kq = K4[idx];
        const float4 vq = V4[idx];
        uint2 qp, kp;
        qp.x = f2bf2(qq.x * qscale, qq.y * qscale);
        qp.y = f2bf2(qq.z * qscale, qq.w * qscale);
        kp.x = f2bf2(kq.x, kq.y);
        kp.y = f2bf2(kq.z, kq.w);
        *(uint2*)&wQs[n * HD + d4] = qp;
        *(uint2*)&wKs[n * HD + d4] = kp;
        sT[(d4 + 0) * KSTR + n] = f2bf(vq.x);
        sT[(d4 + 1) * KSTR + n] = f2bf(vq.y);
        sT[(d4 + 2) * KSTR + n] = f2bf(vq.z);
        sT[(d4 + 3) * KSTR + n] = f2bf(vq.w);
    }
    __syncthreads();
    ushort_t* wVs = wVt + base;   // [d][seq]
    #pragma unroll
    for (int j = 0; j < (BC * HD) / (8 * THREADS); ++j) {   // 2 iters
        const int idx = tid + j * THREADS;
        const int d   = idx >> 3;
        const int k8  = (idx & 7) * 8;
        *(uint4*)&wVs[(size_t)d * SEQ + kv0 + k8] = *(const uint4*)&sT[d * KSTR + k8];
    }
}

// ========================== main flash-attention kernel =========================
__global__ __launch_bounds__(THREADS)
void flash_attn_bf16_wmma(const float* __restrict__ Q,
                          const float* __restrict__ K,
                          const float* __restrict__ V,
                          const ushort_t* __restrict__ wQ,
                          const ushort_t* __restrict__ wK,
                          const ushort_t* __restrict__ wVt,
                          float* __restrict__ Out,
                          int n_qtiles, int use_ws)
{
    __shared__ __align__(16) ushort_t sK [2][BC * KSTR];        // double-buffered
    __shared__ __align__(16) ushort_t sVt[2][HD * KSTR];
    __shared__ __align__(16) float    sP [WAVES * 16 * PSTR];

    const int tid  = threadIdx.x;
    const int wv   = tid >> 5;
    const int lane = tid & 31;
    const int ln   = lane & 15;
    const int half = lane >> 4;

    const int bh    = blockIdx.x / n_qtiles;
    const int qtile = blockIdx.x % n_qtiles;
    const int qbase = qtile * BR + wv * 16;

    const size_t base = (size_t)bh * SEQ * HD;
    const float* Qg = Q + base;
    const float* Kg = K + base;
    const float* Vg = V + base;
    float*       Og = Out + base;
    const ushort_t* k16 = wK  + base;   // bf16 [seq][hd]
    const ushort_t* v16 = wVt + base;   // bf16 [hd][seq]

    // per-thread staging coordinates (uint4 = 8 bf16 granularity)
    const int sn0 = tid >> 3,           sc0 = (tid & 7) * 8;
    const int sn1 = (tid + 256) >> 3,   sc1 = ((tid + 256) & 7) * 8;

    // ---- Q tile (16x64) as two A-frags (bf16, exp2 domain) ----
    Frag16 aQ0, aQ1;
    if (use_ws) {
        const ushort_t* q = wQ + base;
        #pragma unroll
        for (int r = 0; r < 8; ++r) {
            const int kd = ((r >= 4) ? 16 : 0) + 8 * half + 2 * (r & 3);
            aQ0.w[r] = *(const unsigned int*)&q[(size_t)(qbase + ln) * HD + kd];
            aQ1.w[r] = *(const unsigned int*)&q[(size_t)(qbase + ln) * HD + kd + 32];
        }
    } else {
        const float qscale = 0.125f * 1.44269504088896340736f;
        #pragma unroll
        for (int e = 0; e < 16; ++e) {
            const int r = e >> 1, s = e & 1;
            const int kd = ((r >= 4) ? 16 : 0) + 8 * half + 2 * (r & 3) + s;
            aQ0.u[e] = f2bf(Qg[(size_t)(qbase + ln) * HD + kd     ] * qscale);
            aQ1.u[e] = f2bf(Qg[(size_t)(qbase + ln) * HD + kd + 32] * qscale);
        }
    }

    // constant B-frag: column 0 all-ones => extra WMMA accumulates row sums (l)
    Frag16 bOne;
    #pragma unroll
    for (int e = 0; e < 16; ++e) bOne.u[e] = (ln == 0) ? 0x3F80 : 0;

    float mrow[8];
    Frag8f O[4], Ol;
    #pragma unroll
    for (int v = 0; v < 8; ++v) { mrow[v] = -3.0e38f; Ol.f[v] = 0.0f; }
    #pragma unroll
    for (int d = 0; d < 4; ++d)
        #pragma unroll
        for (int v = 0; v < 8; ++v) O[d].f[v] = 0.0f;

    float* sPw = sP + wv * 16 * PSTR;

    // prologue: async-stage tile 0 into buffer 0
    if (use_ws) {
        async_cp_b128(&k16[(size_t)sn0 * HD + sc0],      &sK [0][sn0 * KSTR + sc0]);
        async_cp_b128(&k16[(size_t)sn1 * HD + sc1],      &sK [0][sn1 * KSTR + sc1]);
        async_cp_b128(&v16[(size_t)sn0 * SEQ + sc0],     &sVt[0][sn0 * KSTR + sc0]);
        async_cp_b128(&v16[(size_t)sn1 * SEQ + sc1],     &sVt[0][sn1 * KSTR + sc1]);
    }

    int cur = 0;
    // ================= kv tile loop (32 iters) =================
    for (int kv0 = 0; kv0 < SEQ; kv0 += BC) {
        if (use_ws) {
            wait_async0();        // own async copies for tile kv0 done
            __syncthreads();      // everyone's done; prev buffer free
            if (kv0 + BC < SEQ) { // async-stage next tile into other buffer
                const int nxt = cur ^ 1;
                const size_t kn = (size_t)(kv0 + BC);
                async_cp_b128(&k16[(kn + sn0) * HD + sc0],       &sK [nxt][sn0 * KSTR + sc0]);
                async_cp_b128(&k16[(kn + sn1) * HD + sc1],       &sK [nxt][sn1 * KSTR + sc1]);
                async_cp_b128(&v16[(size_t)sn0 * SEQ + kn + sc0],&sVt[nxt][sn0 * KSTR + sc0]);
                async_cp_b128(&v16[(size_t)sn1 * SEQ + kn + sc1],&sVt[nxt][sn1 * KSTR + sc1]);
            }
        } else {
            __syncthreads();
            const float4* K4 = (const float4*)(Kg + (size_t)kv0 * HD);
            const float4* V4 = (const float4*)(Vg + (size_t)kv0 * HD);
            #pragma unroll
            for (int j = 0; j < (BC * HD) / (4 * THREADS); ++j) {
                const int idx = tid + j * THREADS;
                const int n   = idx >> 4;
                const int d4  = (idx & 15) * 4;
                const float4 kq = K4[idx];
                const float4 vq = V4[idx];
                uint2 kp;
                kp.x = f2bf2(kq.x, kq.y);
                kp.y = f2bf2(kq.z, kq.w);
                *(uint2*)&sK[0][n * KSTR + d4] = kp;
                sVt[0][(d4 + 0) * KSTR + n] = f2bf(vq.x);
                sVt[0][(d4 + 1) * KSTR + n] = f2bf(vq.y);
                sVt[0][(d4 + 2) * KSTR + n] = f2bf(vq.z);
                sVt[0][(d4 + 3) * KSTR + n] = f2bf(vq.w);
            }
            __syncthreads();
        }
        const ushort_t* cK = sK [cur];
        const ushort_t* cV = sVt[cur];

        // ---- S = Q * K^T : 16x64 per wave (four 16x16 frags) ----
        Frag8f S[4];
        #pragma unroll
        for (int nb = 0; nb < 4; ++nb)
            #pragma unroll
            for (int v = 0; v < 8; ++v) S[nb].f[v] = 0.0f;

        #pragma unroll
        for (int nb = 0; nb < 4; ++nb) {
            #pragma unroll
            for (int ks = 0; ks < 2; ++ks) {
                Frag16 bK;   // B 32x16: N = kv row (lane%16), K = ks*32 + 8*half + e (e<8), ks*32 + 16 + 8*half + (e-8) (e>=8)
                const ushort_t* src = &cK[(nb * 16 + ln) * KSTR + ks * 32 + 8 * half];
                bK.q[0] = ((const uint4*)src)[0];
                bK.q[1] = ((const uint4*)src)[2];
                S[nb].v = __builtin_amdgcn_wmma_f32_16x16x32_bf16(
                    false, ks ? aQ1.v : aQ0.v, false, bK.v,
                    (short)0, S[nb].v, false, false);
            }
        }

        // ---- online softmax: DPP row-max, exp2, spill exp(S) to LDS (f32) ----
        float scale[8];
        #pragma unroll
        for (int v = 0; v < 8; ++v) {
            float t = fmaxf(fmaxf(S[0].f[v], S[1].f[v]), fmaxf(S[2].f[v], S[3].f[v]));
            t = row16_max(t);
            const float mnew = fmaxf(mrow[v], t);
            scale[v] = exp2f(mrow[v] - mnew);
            mrow[v]  = mnew;
            #pragma unroll
            for (int nb = 0; nb < 4; ++nb) {
                const float p = exp2f(S[nb].f[v] - mnew);
                sPw[(v + 8 * half) * PSTR + nb * 16 + ln] = p;
            }
        }
        #pragma unroll
        for (int d = 0; d < 4; ++d)
            #pragma unroll
            for (int v = 0; v < 8; ++v) O[d].f[v] *= scale[v];
        #pragma unroll
        for (int v = 0; v < 8; ++v) Ol.f[v] *= scale[v];

        // ---- P: LDS f32 -> bf16 A-frags (packed converts) ----
        Frag16 aP[2];
        #pragma unroll
        for (int ks = 0; ks < 2; ++ks) {
            #pragma unroll
            for (int r = 0; r < 8; ++r) {
                const int kb = ks * 32 + ((r >= 4) ? 16 : 0) + 8 * half + 2 * (r & 3);
                const float2 pr = *(const float2*)&sPw[ln * PSTR + kb];
                aP[ks].w[r] = f2bf2(pr.x, pr.y);
            }
        }

        // ---- O += P * V  and  l += P * ones ----
        #pragma unroll
        for (int db = 0; db < 4; ++db) {
            #pragma unroll
            for (int ks = 0; ks < 2; ++ks) {
                Frag16 bV;  // B 32x16: N = head-dim col, K = ks*32 + 8*half + e (e<8), ks*32 + 16 + 8*half + (e-8) (e>=8)
                const ushort_t* src = &cV[(db * 16 + ln) * KSTR + ks * 32 + 8 * half];
                bV.q[0] = ((const uint4*)src)[0];
                bV.q[1] = ((const uint4*)src)[2];
                O[db].v = __builtin_amdgcn_wmma_f32_16x16x32_bf16(
                    false, aP[ks].v, false, bV.v, (short)0, O[db].v, false, false);
            }
        }
        Ol.v = __builtin_amdgcn_wmma_f32_16x16x32_bf16(
            false, aP[0].v, false, bOne.v, (short)0, Ol.v, false, false);
        Ol.v = __builtin_amdgcn_wmma_f32_16x16x32_bf16(
            false, aP[1].v, false, bOne.v, (short)0, Ol.v, false, false);

        if (use_ws) cur ^= 1;
    }

    // ---- normalize (l lives in lane 0/16 of each half) and store fp32 ----
    #pragma unroll
    for (int v = 0; v < 8; ++v) {
        const float lv  = __shfl(Ol.f[v], 16 * half, 32);
        const float inv = 1.0f / lv;
        const int   row = qbase + v + 8 * half;
        #pragma unroll
        for (int db = 0; db < 4; ++db)
            Og[(size_t)row * HD + db * 16 + ln] = O[db].f[v] * inv;
    }
}

extern "C" void kernel_launch(void* const* d_in, const int* in_sizes, int n_in,
                              void* d_out, int out_size, void* d_ws, size_t ws_size,
                              hipStream_t stream) {
    const float* Q = (const float*)d_in[0];
    const float* K = (const float*)d_in[1];
    const float* V = (const float*)d_in[2];
    float* Out = (float*)d_out;

    const int    bh  = in_sizes[0] / (SEQ * HD);          // batch*heads = 24
    const size_t N   = (size_t)bh * SEQ * HD;
    const int use_ws = (ws_size >= 3 * N * sizeof(ushort_t)) ? 1 : 0;

    ushort_t* wQ  = (ushort_t*)d_ws;
    ushort_t* wK  = wQ + N;
    ushort_t* wVt = wK + N;

    if (use_ws) {
        dim3 cgrid(bh * (SEQ / BC)), cblock(THREADS);
        hipLaunchKernelGGL(convert_qkv, cgrid, cblock, 0, stream, Q, K, V, wQ, wK, wVt);
    }
    const int n_qtiles = SEQ / BR;                        // 16
    dim3 grid(bh * n_qtiles), block(THREADS);
    hipLaunchKernelGGL(flash_attn_bf16_wmma, grid, block, 0, stream,
                       Q, K, V, wQ, wK, wVt, Out, n_qtiles, use_ws);
}
